// MultiHeadMaskedSelfAttention_14113262535375
// MI455X (gfx1250) — matrix operand order repaired and checked by reading, not compiled, not run on hardware
//
#include <hip/hip_runtime.h>

// ---------------------------------------------------------------------------
// MultiHeadMaskedSelfAttention for MI455X (gfx1250, wave32, WMMA)
// B=4, S=2048, D=1024, H=16, HD=64
// Pipeline: x->bf16 ; W->W^T bf16 ; Q/K/V GEMMs (V stored transposed) ;
//           causal flash attention with TRANSPOSED scores (S^T = K @ Q^T,
//           O^T = V^T @ P^T) so softmax stats are per-lane ; out projection.
// Addressing: wave indices forced to SGPRs (readfirstlane) + 32-bit lane
//             offsets so global accesses use saddr+voffset (no 64-bit VGPR
//             address pairs -> no scratch spills in the hot loops).
// ---------------------------------------------------------------------------

typedef __bf16 bf16_t;
typedef __attribute__((ext_vector_type(8)))  __bf16 v8bf;
typedef __attribute__((ext_vector_type(16))) __bf16 v16bf;
typedef __attribute__((ext_vector_type(8)))  float  v8f;

static constexpr int BB = 4;
static constexpr int SS = 2048;
static constexpr int DD = 1024;
static constexpr int HH = 16;
static constexpr int HD = 64;
static constexpr int ROWS = BB * SS;          // 8192 flattened token rows
#define LOG2E 1.4426950408889634f

static __device__ __forceinline__ bf16_t f2bf(float f) { return (bf16_t)f; }

static __device__ __forceinline__ v16bf cat16(v8bf lo, v8bf hi) {
  v16bf r;
#pragma unroll
  for (int i = 0; i < 8; ++i) { r[i] = lo[i]; r[i + 8] = hi[i]; }
  return r;
}
static __device__ __forceinline__ v8f zero8() {
  v8f z;
#pragma unroll
  for (int i = 0; i < 8; ++i) z[i] = 0.0f;
  return z;
}
static __device__ __forceinline__ v8f wmma_bf16(v16bf a, v16bf b, v8f c) {
  // (neg_a, A, neg_b, B, c_mod, C, reuse_a, reuse_b)
  return __builtin_amdgcn_wmma_f32_16x16x32_bf16(false, a, false, b, (short)0, c,
                                                 false, false);
}
static __device__ __forceinline__ v8bf ld8(const bf16_t* p, int eoff) {
  return *(const v8bf*)(p + eoff);
}
static __device__ __forceinline__ v16bf ld16(const bf16_t* p, int eoff) {
  return *(const v16bf*)(p + eoff);
}

// ---------------------------------------------------------------------------
// 1) fp32 -> bf16 elementwise (x), 8 elements/thread, packed 16B stores
// ---------------------------------------------------------------------------
__global__ void convert_x_kernel(const float4* __restrict__ x, v8bf* __restrict__ out) {
  size_t i = (size_t)blockIdx.x * blockDim.x + threadIdx.x;
  float4 a = x[2 * i], b = x[2 * i + 1];
  v8bf o;
  o[0] = f2bf(a.x); o[1] = f2bf(a.y); o[2] = f2bf(a.z); o[3] = f2bf(a.w);
  o[4] = f2bf(b.x); o[5] = f2bf(b.y); o[6] = f2bf(b.z); o[7] = f2bf(b.w);
  out[i] = o;
}

// ---------------------------------------------------------------------------
// 2) W[k][n] fp32 -> WT[n][k] bf16 (tiled LDS transpose, 32x32 tiles)
// ---------------------------------------------------------------------------
__global__ void transpose_w_kernel(const float* __restrict__ W, bf16_t* __restrict__ WT) {
  __shared__ float tile[32][33];
  const int bx = blockIdx.x * 32, by = blockIdx.y * 32;
  const int tx = threadIdx.x, ty = threadIdx.y;   // 32 x 8
#pragma unroll
  for (int i = 0; i < 32; i += 8)
    tile[ty + i][tx] = W[(size_t)(by + ty + i) * DD + bx + tx];
  __syncthreads();
#pragma unroll
  for (int i = 0; i < 32; i += 8)
    WT[(size_t)(bx + ty + i) * DD + by + tx] = f2bf(tile[tx][ty + i]);
}

// ---------------------------------------------------------------------------
// 3) GEMM: C[8192,1024] = A(bf16) @ WT^T + bias. Wave computes 32x64
//    (two 16-row sub-tiles g sharing the four B operands -> 8 wmma / k-chunk).
//    MODE 0: bf16 row-major store, (acc+bias)*scale   (Q with 1/8, K with 1)
//    MODE 1: bf16 transposed store -> VT[b][n][s], packed b128 stores (V)
//    MODE 2: fp32 row-major store                     (final projection)
// ---------------------------------------------------------------------------
template <int MODE>
__global__ void gemm_bf16_kernel(const bf16_t* __restrict__ A,
                                 const bf16_t* __restrict__ WT,
                                 const float* __restrict__ bias,
                                 void* __restrict__ Cout, float scale) {
  const int lane = threadIdx.x & 31;
  const int wave = __builtin_amdgcn_readfirstlane((int)(threadIdx.x >> 5));
  const int lm = lane & 15;
  const int hi = lane >> 4;
  const int mbase = blockIdx.x * 256 + wave * 32;     // SGPR
  const int nbase = blockIdx.y * 64;                  // SGPR

  v8f acc[2][4];
#pragma unroll
  for (int g = 0; g < 2; ++g)
#pragma unroll
    for (int t = 0; t < 4; ++t) acc[g][t] = zero8();

  // Uniform bases + 32-bit lane offsets (saddr + voffset addressing)
  const bf16_t* Abase = A + (size_t)mbase * DD;       // SGPR pair
  const bf16_t* Bbase = WT + (size_t)nbase * DD;      // SGPR pair
  const int aoff = lm * DD + hi * 8;                  // + g*16*DD (const)
  const int boff = lm * DD + hi * 8;                  // + t*16*DD (const)

  for (int kc = 0; kc < DD; kc += 32) {
    v16bf aOp0 = cat16(ld8(Abase, aoff + kc), ld8(Abase, aoff + kc + 16));
    v16bf aOp1 = cat16(ld8(Abase, aoff + 16 * DD + kc),
                       ld8(Abase, aoff + 16 * DD + kc + 16));
#pragma unroll
    for (int t = 0; t < 4; ++t) {
      v16bf bOp = cat16(ld8(Bbase, boff + t * 16 * DD + kc),
                        ld8(Bbase, boff + t * 16 * DD + kc + 16));
      acc[0][t] = wmma_bf16(aOp0, bOp, acc[0][t]);
      acc[1][t] = wmma_bf16(aOp1, bOp, acc[1][t]);
    }
  }

  const float* bb = bias + nbase;                     // SGPR pair
#pragma unroll
  for (int g = 0; g < 2; ++g) {
#pragma unroll
    for (int t = 0; t < 4; ++t) {
      const float bs = bb[t * 16 + lm];
      if constexpr (MODE == 1) {
        // VT[b][n][s]: r -> consecutive s, pack 8 bf16 -> one b128 store
        const int bidx = mbase >> 11;                 // batch (uniform)
        const int s0 = (mbase & (SS - 1)) + 16 * g + 8 * hi;
        bf16_t* vtb = (bf16_t*)Cout + ((size_t)bidx * DD + nbase) * SS;
        v8bf pk;
#pragma unroll
        for (int r = 0; r < 8; ++r) pk[r] = f2bf(acc[g][t][r] + bs);
        *(v8bf*)(vtb + (t * 16 + lm) * SS + s0) = pk;
      } else if constexpr (MODE == 0) {
        bf16_t* ob = (bf16_t*)Cout + (size_t)mbase * DD + nbase;
#pragma unroll
        for (int r = 0; r < 8; ++r)
          ob[(16 * g + r + 8 * hi) * DD + t * 16 + lm] =
              f2bf((acc[g][t][r] + bs) * scale);
      } else {
        float* ob = (float*)Cout + (size_t)mbase * DD + nbase;
#pragma unroll
        for (int r = 0; r < 8; ++r)
          ob[(16 * g + r + 8 * hi) * DD + t * 16 + lm] = acc[g][t][r] + bs;
      }
    }
  }
}

// ---------------------------------------------------------------------------
// 4) Causal flash attention, 32-query tile per wave, TRANSPOSED scores.
//    S^T = K @ Q^T  (M=key, N=query)  -> softmax stats are per-lane scalars
//    O^T = V^T @ P^T (M=d,  N=query)  -> packed b128 output stores
//    4 waves / block; all global traffic via uniform base + 32-bit offset.
// ---------------------------------------------------------------------------
__global__ void __launch_bounds__(128, 1)
flash_attn_kernel(const bf16_t* __restrict__ Q, const bf16_t* __restrict__ Kb,
                  const bf16_t* __restrict__ VT, bf16_t* __restrict__ AO) {
  // per wave, per query sub-tile g: P^T stored as [q(16)][k(32)] bf16
  __shared__ __align__(32) __bf16 pshm[4][2][16 * 32];
  const int lane = threadIdx.x & 31;
  const int wave = __builtin_amdgcn_readfirstlane((int)(threadIdx.x >> 5));
  const int lm = lane & 15;
  const int hi = lane >> 4;

  const int tile = blockIdx.x * 4 + wave;  // 4096 tiles of 32 queries (SGPR)
  const int qt = tile & 63;                // 64 query tiles per (b,h)
  const int h = (tile >> 6) & 15;
  const int b = tile >> 10;
  const int qbase = qt * 32;

  // Uniform bases (SGPR pairs)
  const bf16_t* Qbase = Q + (size_t)(b * SS + qbase) * DD + h * HD;
  const bf16_t* Kbase = Kb + (size_t)(b * SS) * DD + h * HD;
  const bf16_t* Vbase = VT + (size_t)(b * DD + h * HD) * SS;
  bf16_t* Obase = AO + (size_t)(b * SS + qbase) * DD + h * HD;

  // Q as B operands (lane = query column, two runs of 8 d (8hi.., 16+8hi..) per half-wave)
  v16bf bQ[2][2];
  {
    const int qo = lm * DD + hi * 8;       // + g*16*DD, + kk*32 (consts)
#pragma unroll
    for (int g = 0; g < 2; ++g) {
      bQ[g][0] = cat16(ld8(Qbase, qo + g * 16 * DD), ld8(Qbase, qo + g * 16 * DD + 16));
      bQ[g][1] = cat16(ld8(Qbase, qo + g * 16 * DD + 32), ld8(Qbase, qo + g * 16 * DD + 48));
    }
  }

  float m[2] = {-1e30f, -1e30f}, l[2] = {0.0f, 0.0f};
  v8f o[2][4];
#pragma unroll
  for (int g = 0; g < 2; ++g)
#pragma unroll
    for (int t = 0; t < 4; ++t) o[g][t] = zero8();

  const int koff = lm * DD + hi * 8;       // K lane offset (+kt*16*DD, +16,+32,+48)
  const int voff = lm * SS + hi * 8;       // V lane offset (+t*16*SS, +16)
  const int poff_st = lm * 32 + 16 * hi;   // P^T LDS store offset
  const int poff_ld = lm * 32 + hi * 16;   // P^T LDS load offset (B operand)

  for (int chunk = 0; chunk <= qbase; chunk += 32) {
    // ---- K as A operands (shared by both query sub-tiles)
    const int kc = koff + chunk * DD;      // chunk*DD is uniform
    v16bf aK[2][2];
#pragma unroll
    for (int kt = 0; kt < 2; ++kt) {
      aK[kt][0] = cat16(ld8(Kbase, kc + kt * 16 * DD),
                        ld8(Kbase, kc + kt * 16 * DD + 16));
      aK[kt][1] = cat16(ld8(Kbase, kc + kt * 16 * DD + 32),
                        ld8(Kbase, kc + kt * 16 * DD + 48));
    }

    const bool need_mask = (chunk == qbase);   // only the diagonal chunk

#pragma unroll
    for (int g = 0; g < 2; ++g) {
      // ---- S^T tiles: [key 16 x query 16], chained over head dim (K=32 x2)
      v8f st[2];
#pragma unroll
      for (int kt = 0; kt < 2; ++kt) {
        v8f s = zero8();
        s = wmma_bf16(aK[kt][0], bQ[g][0], s);
        s = wmma_bf16(aK[kt][1], bQ[g][1], s);
        st[kt] = s;
      }

      if (need_mask) {
        const int q = qbase + 16 * g + lm;
#pragma unroll
        for (int kt = 0; kt < 2; ++kt)
#pragma unroll
          for (int r = 0; r < 8; ++r) {
            const int key = chunk + kt * 16 + r + 8 * hi;
            if (key > q) st[kt][r] = -1e30f;
          }
      }

      // ---- per-lane softmax stats (lane owns query column lm; one
      //      cross-half shuffle combines the two 8-key halves)
      float mx = st[0][0];
#pragma unroll
      for (int kt = 0; kt < 2; ++kt)
#pragma unroll
        for (int r = 0; r < 8; ++r) mx = fmaxf(mx, st[kt][r]);
      mx = fmaxf(mx, __shfl_xor(mx, 16, 32));
      const float mnew = fmaxf(m[g], mx);
      const float c = __builtin_exp2f((m[g] - mnew) * LOG2E);
      m[g] = mnew;

      float ps = 0.0f;
      v8bf pk[2];
#pragma unroll
      for (int kt = 0; kt < 2; ++kt)
#pragma unroll
        for (int r = 0; r < 8; ++r) {
          const float p = __builtin_exp2f((st[kt][r] - mnew) * LOG2E);
          ps += p;
          pk[kt][r] = f2bf(p);
        }
      ps += __shfl_xor(ps, 16, 32);
      l[g] = l[g] * c + ps;
#pragma unroll
      for (int t = 0; t < 4; ++t) o[g][t] *= c;

      // ---- P^T to LDS [q][k]; lane holds 8 consecutive k -> b128 stores
      __bf16* pw = &pshm[wave][g][poff_st];
      *(v8bf*)(pw) = pk[0];        // k = 8hi + r      (kt = 0)
      *(v8bf*)(pw + 8) = pk[1];    // k = 16 + 8hi + r (kt = 1)
    }

    asm volatile("" ::: "memory");  // intra-wave DS ops are in-order

    // ---- O^T += V^T @ P^T : A = V^T tile (d x 32k), B = P^T (32k x q)
    v16bf bP0 = *(const v16bf*)(&pshm[wave][0][poff_ld]);
    v16bf bP1 = *(const v16bf*)(&pshm[wave][1][poff_ld]);
    const int vc = voff + chunk;           // chunk is uniform
#pragma unroll
    for (int t = 0; t < 4; ++t) {
      v16bf aV = cat16(ld8(Vbase, vc + t * 16 * SS),
                       ld8(Vbase, vc + t * 16 * SS + 16));
      o[0][t] = wmma_bf16(aV, bP0, o[0][t]);
      o[1][t] = wmma_bf16(aV, bP1, o[1][t]);
    }
    asm volatile("" ::: "memory");
  }

  // ---- normalize, pack 8 consecutive d per lane -> b128 stores
#pragma unroll
  for (int g = 0; g < 2; ++g) {
    const float inv = 1.0f / l[g];
    const int oo = (16 * g + lm) * DD + 8 * hi;  // + t*16 (const)
#pragma unroll
    for (int t = 0; t < 4; ++t) {
      v8bf pk;
#pragma unroll
      for (int r = 0; r < 8; ++r) pk[r] = f2bf(o[g][t][r] * inv);
      *(v8bf*)(Obase + oo + t * 16) = pk;
    }
  }
}

// ---------------------------------------------------------------------------
// Host-side launcher
// ---------------------------------------------------------------------------
extern "C" void kernel_launch(void* const* d_in, const int* in_sizes, int n_in,
                              void* d_out, int out_size, void* d_ws, size_t ws_size,
                              hipStream_t stream) {
  const float* x  = (const float*)d_in[0];
  const float* Wq = (const float*)d_in[1];
  const float* bq = (const float*)d_in[2];
  const float* Wk = (const float*)d_in[3];
  const float* bk = (const float*)d_in[4];
  const float* Wv = (const float*)d_in[5];
  const float* bv = (const float*)d_in[6];
  const float* Wo = (const float*)d_in[7];
  const float* bo = (const float*)d_in[8];
  float* out = (float*)d_out;

  // Workspace layout (bf16 elements): ~92 MB total
  const size_t XN = (size_t)ROWS * DD;        // 8,388,608
  const size_t WN = (size_t)DD * DD;          // 1,048,576
  bf16_t* xb  = (bf16_t*)d_ws;
  bf16_t* WqT = xb + XN;
  bf16_t* WkT = WqT + WN;
  bf16_t* WvT = WkT + WN;
  bf16_t* WoT = WvT + WN;
  bf16_t* Qb  = WoT + WN;
  bf16_t* Kbf = Qb + XN;
  bf16_t* VTb = Kbf + XN;
  bf16_t* AO  = VTb + XN;

  // 1) convert x to bf16 (8 elems/thread)
  convert_x_kernel<<<(unsigned)(XN / 8 / 256), 256, 0, stream>>>(
      (const float4*)x, (v8bf*)xb);

  // 2) transpose + convert weights
  dim3 tg(DD / 32, DD / 32), tb(32, 8);
  transpose_w_kernel<<<tg, tb, 0, stream>>>(Wq, WqT);
  transpose_w_kernel<<<tg, tb, 0, stream>>>(Wk, WkT);
  transpose_w_kernel<<<tg, tb, 0, stream>>>(Wv, WvT);
  transpose_w_kernel<<<tg, tb, 0, stream>>>(Wo, WoT);

  // 3) projections: Q (scaled by 1/sqrt(HD)=0.125), K, V (stored transposed)
  dim3 gg(ROWS / 256, DD / 64);
  gemm_bf16_kernel<0><<<gg, 256, 0, stream>>>(xb, WqT, bq, Qb, 0.125f);
  gemm_bf16_kernel<0><<<gg, 256, 0, stream>>>(xb, WkT, bk, Kbf, 1.0f);
  gemm_bf16_kernel<1><<<gg, 256, 0, stream>>>(xb, WvT, bv, VTb, 1.0f);

  // 4) causal flash attention: 4096 wave-tiles (32 queries), 4 waves/block
  flash_attn_kernel<<<(BB * HH * (SS / 32)) / 4, 128, 0, stream>>>(Qb, Kbf, VTb, AO);

  // 5) output projection -> fp32
  gemm_bf16_kernel<2><<<gg, 256, 0, stream>>>(AO, WoT, bo, out, 1.0f);
}
